// MAN_41257455845894
// MI455X (gfx1250) — matrix operand order repaired and checked by reading, not compiled, not run on hardware
//
#include <hip/hip_runtime.h>
#include <hip/hip_bf16.h>

typedef __attribute__((ext_vector_type(16))) _Float16 v16h;
typedef __attribute__((ext_vector_type(8)))  float    v8f;

#define BB     128
#define TT     1024
#define INSZ   256
#define HID    256
#define MEMN   128
#define FOURH  1024
#define KTOT   512          // INSZ + HID
#define NWG    128
#define GATEV  0.4013123400f   // sigmoid(-0.4)
#define GAMMAV 0.3f

// fragment-buffer geometry
#define NTILE  64            // 4H / 16 column tiles
#define KCHUNK 16            // KTOT / 32 k-chunks
#define RTILE  8             // B / 16 row tiles
#define FRAGSZ (32*16)       // f16 per fragment (32 lanes x 16 elems)

// ---- workspace layout (bytes) ----
#define OFF_WF     0                                       // [NTILE][KCHUNK][32][16] f16 = 1 MB
#define OFF_XF     (OFF_WF + NTILE*KCHUNK*FRAGSZ*2)        // 2 x [RTILE][8][32][16] f16 (ping-pong)
#define OFF_HF     (OFF_XF + 2*RTILE*8*FRAGSZ*2)           // [RTILE][8][32][16] f16
#define OFF_BIAS   (OFF_HF + RTILE*8*FRAGSZ*2)             // [FOURH] f32 (b_ih+b_hh)
#define OFF_H      (OFF_BIAS + FOURH*4)                    // [BB][HID] f32
#define OFF_C      (OFF_H    + BB*HID*4)
#define OFF_M      (OFF_C    + BB*HID*4)                   // [MEMN][HID] f32
#define OFF_MNORM  (OFF_M    + MEMN*HID*4)                 // [MEMN]
#define OFF_HNORM  (OFF_MNORM+ MEMN*4)                     // [BB]
#define OFF_WU     (OFF_HNORM+ BB*4)                       // [BB][MEMN]
#define OFF_WR0    (OFF_WU   + BB*MEMN*4)                  // ping
#define OFF_WR1    (OFF_WR0  + BB*MEMN*4)                  // pong
#define OFF_WW     (OFF_WR1  + BB*MEMN*4)                  // [BB][MEMN]
#define OFF_Z      (OFF_WW   + BB*MEMN*4)                  // [BB][FOURH] f32
#define OFF_BAR    (OFF_Z    + BB*FOURH*4)                 // 2 ints
#define WS_END     (OFF_BAR + 16)

__device__ __forceinline__ float sigf(float x) { return 1.0f / (1.0f + __expf(-x)); }

// A-matrix 16x32 f16 layout (ISA 7.12.2): given local k (0..31) -> lane group g, elem e
__device__ __forceinline__ void a_pos(int kl, int& g, int& e) {
    if (kl < 16) { g = kl >> 3; e = kl & 7; }
    else { int r = kl - 16; g = r >> 3; e = 8 + (r & 7); }
}

// -------- grid-wide barrier (persistent kernel) --------
__device__ void gridbar(int* cnt, volatile int* gen) {
    __syncthreads();
    if (threadIdx.x == 0) {
        __threadfence();
        int g = *gen;
        if (atomicAdd(cnt, 1) == NWG - 1) {
            *cnt = 0;
            __threadfence();
            atomicAdd((int*)gen, 1);
        } else {
            while (*gen == g) { __builtin_amdgcn_s_sleep(2); }
        }
        __threadfence();
    }
    __syncthreads();
}

// -------- init: zero mutable state (HF, norms, state, barrier) --------
__global__ void MAN_prep_state(char* ws) {
    size_t n = (size_t)(WS_END - OFF_HF) / 4;
    size_t i = (size_t)blockIdx.x * blockDim.x + threadIdx.x;
    if (i < n) ((int*)(ws + OFF_HF))[i] = 0;
}

// -------- pack weights into per-lane B-fragment layout + fuse biases --------
__global__ void MAN_prep_weights(const float* __restrict__ W_ih,
                                 const float* __restrict__ W_hh,
                                 const float* __restrict__ b_ih,
                                 const float* __restrict__ b_hh,
                                 char* ws) {
    _Float16* wf   = (_Float16*)(ws + OFF_WF);
    float*    bias = (float*)(ws + OFF_BIAS);
    size_t i = (size_t)blockIdx.x * blockDim.x + threadIdx.x;
    if (i < (size_t)NTILE * KCHUNK * FRAGSZ) {
        int e    = (int)(i & 15);
        int lane = (int)((i >> 4) & 31);
        int kc   = (int)((i >> 9) & 15);
        int nc   = (int)(i >> 13);
        int n = nc * 16 + (lane & 15);
        int k = kc * 32 + ((e < 8) ? (lane >> 4) * 8 + e : 16 + (lane >> 4) * 8 + (e - 8));     // B layout: table order, matches A (a_pos)
        float v = (k < INSZ) ? W_ih[(size_t)n * INSZ + k]
                             : W_hh[(size_t)n * HID + (k - INSZ)];
        wf[i] = (_Float16)v;
    }
    if (i < FOURH) bias[i] = b_ih[i] + b_hh[i];
}

// -------- pack x_0 into A-fragment layout (ping buffer 0) --------
__global__ void MAN_prep_x0(const float* __restrict__ X, char* ws) {
    _Float16* xf0 = (_Float16*)(ws + OFF_XF);
    int i = blockIdx.x * blockDim.x + threadIdx.x;   // BB*INSZ
    if (i < BB * INSZ) {
        int b = i >> 8, j = i & 255;
        int kc = j >> 5, kl = j & 31, g, e;
        a_pos(kl, g, e);
        int lane = (b & 15) + 16 * g;
        xf0[(((b >> 4) * 8 + kc) * 32 + lane) * 16 + e] =
            (_Float16)X[((size_t)b * TT + 0) * INSZ + j];
    }
}

// -------- persistent scan kernel --------
__global__ __launch_bounds__(256, 1)
void MAN_recur(const float* __restrict__ X, float* __restrict__ out, char* ws) {
    _Float16* wf  = (_Float16*)(ws + OFF_WF);
    _Float16* xfA = (_Float16*)(ws + OFF_XF);
    _Float16* xfB = xfA + RTILE * 8 * FRAGSZ;
    _Float16* hf  = (_Float16*)(ws + OFF_HF);
    float* bias  = (float*)(ws + OFF_BIAS);
    float* h     = (float*)(ws + OFF_H);
    float* c     = (float*)(ws + OFF_C);
    float* Mm    = (float*)(ws + OFF_M);
    float* Mnorm = (float*)(ws + OFF_MNORM);
    float* hnorm = (float*)(ws + OFF_HNORM);
    float* wu    = (float*)(ws + OFF_WU);
    float* wrbuf0= (float*)(ws + OFF_WR0);
    float* wrbuf1= (float*)(ws + OFF_WR1);
    float* wwb   = (float*)(ws + OFF_WW);
    float* z     = (float*)(ws + OFF_Z);
    int*   bcnt  = (int*)(ws + OFF_BAR);
    int*   bgen  = bcnt + 1;

    __shared__ float sm[256];
    __shared__ float sv[MEMN];
    __shared__ float red[128];
    __shared__ int   redi[128];

    const int tid   = threadIdx.x;
    const int wave  = tid >> 5;
    const int lane  = tid & 31;
    const int lhalf = lane & 15;
    const int lg    = lane >> 4;
    const int blk   = blockIdx.x;

    for (int t = 0; t < TT; ++t) {
        const _Float16* xcur  = (t & 1) ? xfB : xfA;
        _Float16*       xnext = (t & 1) ? xfA : xfB;

        // ===== P1: z = [x_t | h] @ Wcat^T + bias  (pre-swizzled WMMA frags) =====
        {
            int tile = blk * 8 + wave;            // 8 row-tiles x 64 col-tiles
            if (tile < RTILE * NTILE) {
                int tr = tile >> 6;
                int nc = tile & 63;
                v8f acc;
                float bv = bias[nc * 16 + lhalf];
                #pragma unroll
                for (int r = 0; r < 8; ++r) acc[r] = bv;

                const v16h* ax = (const v16h*)(xcur + ((tr * 8) * 32 + lane) * 16);
                const v16h* ah = (const v16h*)(hf   + ((tr * 8) * 32 + lane) * 16);
                const v16h* bw = (const v16h*)(wf   + ((nc * 16) * 32 + lane) * 16);

                #pragma unroll
                for (int kc = 0; kc < 8; ++kc) {          // K = 0..255 (x part)
                    v16h a  = ax[kc * 32];
                    v16h bf = bw[kc * 32];
                    acc = __builtin_amdgcn_wmma_f32_16x16x32_f16(
                        false, a, false, bf, (short)0, acc, false, false);
                }
                #pragma unroll
                for (int kc = 0; kc < 8; ++kc) {          // K = 256..511 (h part)
                    v16h a  = ah[kc * 32];
                    v16h bf = bw[(8 + kc) * 32];
                    acc = __builtin_amdgcn_wmma_f32_16x16x32_f16(
                        false, a, false, bf, (short)0, acc, false, false);
                }
                #pragma unroll
                for (int r = 0; r < 8; ++r) {
                    int m = tr * 16 + r + 8 * lg;
                    z[(size_t)m * FOURH + nc * 16 + lhalf] = acc[r];
                }
            }
        }
        gridbar(bcnt, bgen);

        // ===== P2: LSTM pointwise, ||h||, pack h-frag + next-x-frag =====
        {
            int b = blk, j = tid;
            float iv = z[(size_t)b * FOURH + j];
            float fv = z[(size_t)b * FOURH + 256 + j];
            float gv = z[(size_t)b * FOURH + 512 + j];
            float ov = z[(size_t)b * FOURH + 768 + j];
            float cc = sigf(fv) * c[b * HID + j] + sigf(iv) * tanhf(gv);
            float hh = sigf(ov) * tanhf(cc);
            c[b * HID + j] = cc;
            h[b * HID + j] = hh;
            out[((size_t)b * TT + t) * 512 + j] = hh;        // out[:, t, 0:256] = h

            int kc = j >> 5, kl = j & 31, g, e;
            a_pos(kl, g, e);
            int lp  = (b & 15) + 16 * g;
            int trp = b >> 4;
            hf[((trp * 8 + kc) * 32 + lp) * 16 + e] = (_Float16)hh;
            if (t + 1 < TT)                                   // stage x_{t+1}
                xnext[((trp * 8 + kc) * 32 + lp) * 16 + e] =
                    (_Float16)X[((size_t)b * TT + (t + 1)) * INSZ + j];

            sm[j] = hh * hh;
            __syncthreads();
            for (int s = 128; s > 0; s >>= 1) {
                if (j < s) sm[j] += sm[j + s];
                __syncthreads();
            }
            if (j == 0) hnorm[b] = sqrtf(sm[0]);
        }
        gridbar(bcnt, bgen);

        // ===== P3: read head (cosine softmax), LRU argmin, w_w =====
        {
            int b = blk, j = tid;
            float* wrcur  = (t & 1) ? wrbuf1 : wrbuf0;
            float* wrprev = (t & 1) ? wrbuf0 : wrbuf1;

            if (j < MEMN) {
                const float* hb = h + b * HID;
                const float* Ms = Mm + j * HID;
                float dot = 0.f;
                #pragma unroll 8
                for (int q = 0; q < HID; ++q) dot += hb[q] * Ms[q];
                sv[j] = dot / (hnorm[b] * Mnorm[j] + 1e-8f);
            }
            __syncthreads();
            if (j < 128) red[j] = sv[j];
            __syncthreads();
            for (int s = 64; s > 0; s >>= 1) {               // max
                if (j < s) red[j] = fmaxf(red[j], red[j + s]);
                __syncthreads();
            }
            float mx = red[0];
            __syncthreads();
            if (j < 128) { float ev = __expf(sv[j] - mx); sv[j] = ev; red[j] = ev; }
            __syncthreads();
            for (int s = 64; s > 0; s >>= 1) {               // sum
                if (j < s) red[j] += red[j + s];
                __syncthreads();
            }
            float ssum = red[0];
            __syncthreads();
            if (j < 128) {
                float w = sv[j] / ssum;
                sv[j] = w;
                wrcur[b * MEMN + j] = w;
                red[j] = wu[b * MEMN + j];                   // argmin(w_u) (pre-update)
                redi[j] = j;
            }
            __syncthreads();
            for (int s = 64; s > 0; s >>= 1) {
                if (j < s && red[j + s] < red[j]) { red[j] = red[j + s]; redi[j] = redi[j + s]; }
                __syncthreads();
            }
            int imin = redi[0];
            if (j < 128)
                wwb[b * MEMN + j] = GATEV * wrprev[b * MEMN + j]
                                  + ((j == imin) ? (1.0f - GATEV) : 0.0f);
            __syncthreads();
            float rd = 0.f;                                   // read = w_r @ M
            #pragma unroll 8
            for (int s = 0; s < MEMN; ++s) rd += sv[s] * Mm[s * HID + j];
            out[((size_t)b * TT + t) * 512 + 256 + j] = rd;
        }
        gridbar(bcnt, bgen);

        // ===== P4: M += w_w^T h, row l2norm; w_u update =====
        {
            int s = blk, j = tid;
            float* wrcur = (t & 1) ? wrbuf1 : wrbuf0;
            float acc = Mm[s * HID + j];
            #pragma unroll 8
            for (int b = 0; b < BB; ++b) acc += wwb[b * MEMN + s] * h[b * HID + j];
            sm[j] = acc * acc;
            __syncthreads();
            for (int q = 128; q > 0; q >>= 1) {
                if (j < q) sm[j] += sm[j + q];
                __syncthreads();
            }
            float nrm = sqrtf(sm[0]);
            float inv = 1.0f / fmaxf(nrm, 1e-12f);
            Mm[s * HID + j] = acc * inv;
            if (j == 0) Mnorm[s] = nrm * inv;                 // ||l2norm(row)||
            __syncthreads();
            float v = 0.f;                                     // w_u row (batch row s)
            if (j < MEMN)
                v = GAMMAV * wu[s * MEMN + j] + wrcur[s * MEMN + j] + wwb[s * MEMN + j];
            sm[j] = (j < MEMN) ? v * v : 0.f;
            __syncthreads();
            for (int q = 128; q > 0; q >>= 1) {
                if (j < q) sm[j] += sm[j + q];
                __syncthreads();
            }
            float un = sqrtf(sm[0]);
            if (j < MEMN) wu[s * MEMN + j] = v / fmaxf(un, 1e-12f);
        }
        gridbar(bcnt, bgen);
    }
}

extern "C" void kernel_launch(void* const* d_in, const int* in_sizes, int n_in,
                              void* d_out, int out_size, void* d_ws, size_t ws_size,
                              hipStream_t stream) {
    (void)in_sizes; (void)n_in; (void)out_size; (void)ws_size;
    const float* X    = (const float*)d_in[0];
    const float* W_ih = (const float*)d_in[1];
    const float* W_hh = (const float*)d_in[2];
    const float* b_ih = (const float*)d_in[3];
    const float* b_hh = (const float*)d_in[4];
    float* out = (float*)d_out;
    char*  ws  = (char*)d_ws;

    size_t zero_words = (size_t)(WS_END - OFF_HF) / 4;
    MAN_prep_state<<<(unsigned)((zero_words + 255) / 256), 256, 0, stream>>>(ws);
    MAN_prep_weights<<<(NTILE * KCHUNK * FRAGSZ + 255) / 256, 256, 0, stream>>>(
        W_ih, W_hh, b_ih, b_hh, ws);
    MAN_prep_x0<<<(BB * INSZ + 255) / 256, 256, 0, stream>>>(X, ws);
    MAN_recur<<<NWG, 256, 0, stream>>>(X, out, ws);
}
